// MyRNN_67388036874303
// MI455X (gfx1250) — matrix operand order repaired and checked by reading, not compiled, not run on hardware
//
#include <hip/hip_runtime.h>
#include <hip/hip_bf16.h>
#include <math.h>

// ---------------------------------------------------------------------------
// MyRNN on MI455X (gfx1250): bf16 WMMA recurrent GEMMs, fp32 accumulate.
//
// Reference: h0 = tanh(x_t@k0 + h@rk0 + b0); h1 = tanh(h0@k1 + h@rk1 + b1);
//            h = h1 (both cells recur on h1 due to the source quirk).
// B=128, T=80, E=100, U=512, V=10000.
// ---------------------------------------------------------------------------

#define B_  128
#define T_  80
#define E_  100
#define U_  512

typedef __attribute__((ext_vector_type(16))) __bf16 v16bf;
typedef __attribute__((ext_vector_type(8)))  __bf16 bf16x8;
typedef __attribute__((ext_vector_type(8)))  float  v8f;

// Hardware v_tanh_f32 when the toolchain exposes it (gfx1250 TRANS op),
// OCML fallback otherwise.
__device__ __forceinline__ float fast_tanh(float x) {
#if defined(__has_builtin)
#if __has_builtin(__builtin_amdgcn_tanhf)
    return __builtin_amdgcn_tanhf(x);
#else
    return tanhf(x);
#endif
#else
    return tanhf(x);
#endif
}

// ---------------------------------------------------------------------------
// One-time prep kernels (off the critical path)
// ---------------------------------------------------------------------------

// Transpose 512x512 fp32 weight -> bf16 [n][k] so WMMA B-fragments are
// contiguous per-lane 32B loads.
__global__ void transpose_w_bf16(const float* __restrict__ src,
                                 __bf16* __restrict__ dst) {
    int idx = blockIdx.x * blockDim.x + threadIdx.x;   // 0 .. 512*512-1
    int n = idx >> 9;
    int k = idx & 511;
    dst[n * U_ + k] = (__bf16)src[k * U_ + n];
}

// Transpose k0 [E,U] fp32 -> k0t [U,E] fp32 (for coalesced xproj dot products)
__global__ void transpose_k0(const float* __restrict__ k0,
                             float* __restrict__ k0t) {
    int idx = blockIdx.x * blockDim.x + threadIdx.x;   // 0 .. U_*E_-1
    if (idx >= U_ * E_) return;
    int u = idx / E_;
    int e = idx - u * E_;
    k0t[u * E_ + e] = k0[e * U_ + u];
}

__global__ void zero_bf16(__bf16* __restrict__ p, int n) {
    int idx = blockIdx.x * blockDim.x + threadIdx.x;
    if (idx < n) p[idx] = (__bf16)0.0f;
}

// xproj[t][b][u] = emb[tokens[b,t]] . k0[:,u] + b0[u]   (fp32, 25x float4)
__global__ void xproj_kernel(const int* __restrict__ tokens,
                             const float* __restrict__ emb,
                             const float* __restrict__ k0t,
                             const float* __restrict__ b0,
                             float* __restrict__ xproj) {
    int idx = blockIdx.x * blockDim.x + threadIdx.x;   // T*B*U
    if (idx >= T_ * B_ * U_) return;
    int u  = idx & (U_ - 1);
    int tb = idx >> 9;            // t*B + b
    int b  = tb & (B_ - 1);
    int t  = tb >> 7;
    int tok = tokens[b * T_ + t];
    const float4* er = (const float4*)(emb + (size_t)tok * E_);  // 400B rows, 16B aligned
    const float4* kr = (const float4*)(k0t + (size_t)u * E_);
    float s = 0.0f;
#pragma unroll
    for (int i = 0; i < 25; ++i) {
        float4 e4 = er[i];
        float4 k4 = kr[i];
        s += e4.x * k4.x + e4.y * k4.y + e4.z * k4.z + e4.w * k4.w;
    }
    xproj[idx] = s + b0[u];
}

// ---------------------------------------------------------------------------
// WMMA fragment helpers (wave32; A and B share the contiguous k = (L>>4)*16 + i order)
//   A (16x32 bf16): lane L: m = L&15, koff = (L>>4)*16
//       a[i]   = A[m][kt*32 + koff + i]        i=0..7   (16B contiguous), koff = (L>>4)*16
//       a[8+i] = A[m][kt*32 + koff + 8 + i]    i=0..7   (16B contiguous)
//   B (32x16 bf16, from pre-transposed Wt[n][k]): lane L: n = L&15, koff2=(L>>4)*16
//       b[i]   = Wt[n][kt*32 + koff2 + i]      i=0..15  (32B contiguous)
//   C/D (16x16 f32): lane L: col = L&15, row = (L>>4)*8 + j for c[j]
// ---------------------------------------------------------------------------

__device__ __forceinline__ v16bf load_a_frag(const __bf16* __restrict__ arow,
                                             int kbase, int koffA) {
    bf16x8 lo = *(const bf16x8*)(arow + kbase + koffA);
    bf16x8 hi = *(const bf16x8*)(arow + kbase + koffA + 8);
    v16bf a;
#pragma unroll
    for (int i = 0; i < 8; ++i) { a[i] = lo[i]; a[8 + i] = hi[i]; }
    return a;
}

// 256 independent 16x16 output tiles; spread across 64 WGs x 4 waves so 64
// WGPs work in parallel (latency matters: these sit on the sequential chain).
#define CELL_GRID   64
#define CELL_BLOCK  128

// ---------------------------------------------------------------------------
// Cell 0: h0 = tanh(xproj_t + h @ rk0)
// ---------------------------------------------------------------------------
__global__ __launch_bounds__(CELL_BLOCK, 1)
void rnn_cell0(const __bf16* __restrict__ hb,      // [B,U] bf16 (h_{t-1})
               const __bf16* __restrict__ rk0t,    // [U_out][U_in] bf16
               const float*  __restrict__ xproj_t, // [B,U] fp32 (x_t@k0 + b0)
               __bf16* __restrict__ h0b)           // out [B,U] bf16
{
    const int lane = threadIdx.x & 31;
    const int tile = blockIdx.x * (CELL_BLOCK >> 5) + (threadIdx.x >> 5);
    const int m0 = (tile >> 5) << 4;   // 8 M strips
    const int n0 = (tile & 31) << 4;   // 32 N tiles
    const int l15 = lane & 15;
    const int hi  = lane >> 4;
    const int mo  = hi << 3;
    const int koffA = hi << 4;
    const int koffB = hi << 4;

    v8f c;
#pragma unroll
    for (int j = 0; j < 8; ++j)
        c[j] = xproj_t[(m0 + mo + j) * U_ + n0 + l15];

    const __bf16* arow = hb   + (size_t)(m0 + l15) * U_;
    const __bf16* brow = rk0t + (size_t)(n0 + l15) * U_;

#pragma unroll
    for (int kt = 0; kt < U_ / 32; ++kt) {
        const int kb = kt * 32;
        v16bf a = load_a_frag(arow, kb, koffA);
        v16bf b = *(const v16bf*)(brow + kb + koffB);
        c = __builtin_amdgcn_wmma_f32_16x16x32_bf16(
                false, a, false, b, (short)0, c, false, false);
    }

#pragma unroll
    for (int j = 0; j < 8; ++j)
        h0b[(m0 + mo + j) * U_ + n0 + l15] = (__bf16)fast_tanh(c[j]);
}

// ---------------------------------------------------------------------------
// Cell 1: h1 = tanh(h0 @ k1 + h @ rk1 + b1)   (writes the other h buffer)
// ---------------------------------------------------------------------------
__global__ __launch_bounds__(CELL_BLOCK, 1)
void rnn_cell1(const __bf16* __restrict__ h0b,   // [B,U] bf16
               const __bf16* __restrict__ hb_in, // [B,U] bf16 (h_{t-1})
               const __bf16* __restrict__ k1t,   // [U][U] bf16 transposed
               const __bf16* __restrict__ rk1t,  // [U][U] bf16 transposed
               const float*  __restrict__ b1,    // [U]
               __bf16* __restrict__ hb_out)      // [B,U] bf16 (h_t)
{
    const int lane = threadIdx.x & 31;
    const int tile = blockIdx.x * (CELL_BLOCK >> 5) + (threadIdx.x >> 5);
    const int m0 = (tile >> 5) << 4;
    const int n0 = (tile & 31) << 4;
    const int l15 = lane & 15;
    const int hi  = lane >> 4;
    const int mo  = hi << 3;
    const int koffA = hi << 4;
    const int koffB = hi << 4;

    const float bias = b1[n0 + l15];
    v8f c;
#pragma unroll
    for (int j = 0; j < 8; ++j) c[j] = bias;

    const __bf16* arow0 = h0b   + (size_t)(m0 + l15) * U_;
    const __bf16* arow1 = hb_in + (size_t)(m0 + l15) * U_;
    const __bf16* brow0 = k1t   + (size_t)(n0 + l15) * U_;
    const __bf16* brow1 = rk1t  + (size_t)(n0 + l15) * U_;

#pragma unroll
    for (int kt = 0; kt < U_ / 32; ++kt) {
        const int kb = kt * 32;
        v16bf a0 = load_a_frag(arow0, kb, koffA);
        v16bf b0v = *(const v16bf*)(brow0 + kb + koffB);
        c = __builtin_amdgcn_wmma_f32_16x16x32_bf16(
                false, a0, false, b0v, (short)0, c, false, false);
        v16bf a1 = load_a_frag(arow1, kb, koffA);
        v16bf b1v = *(const v16bf*)(brow1 + kb + koffB);
        c = __builtin_amdgcn_wmma_f32_16x16x32_bf16(
                false, a1, false, b1v, (short)0, c, false, false);
    }

#pragma unroll
    for (int j = 0; j < 8; ++j)
        hb_out[(m0 + mo + j) * U_ + n0 + l15] = (__bf16)fast_tanh(c[j]);
}

// ---------------------------------------------------------------------------
// Output head: out[b] = sigmoid(h_T[b] . wd + bd)
// ---------------------------------------------------------------------------
__global__ void logits_kernel(const __bf16* __restrict__ h,
                              const float* __restrict__ wd,
                              const float* __restrict__ bd,
                              float* __restrict__ out) {
    int b = threadIdx.x;   // 0..127
    float s = 0.0f;
    for (int u = 0; u < U_; ++u)
        s += (float)h[b * U_ + u] * wd[u];
    s += bd[0];
    out[b] = 1.0f / (1.0f + expf(-s));
}

// ---------------------------------------------------------------------------
// Host launcher
// ---------------------------------------------------------------------------
extern "C" void kernel_launch(void* const* d_in, const int* in_sizes, int n_in,
                              void* d_out, int out_size, void* d_ws, size_t ws_size,
                              hipStream_t stream) {
    const int*   tokens = (const int*)  d_in[0];   // [B,T]
    const float* emb    = (const float*)d_in[1];   // [V,E]
    const float* k0     = (const float*)d_in[2];   // [E,U]
    const float* rk0    = (const float*)d_in[3];   // [U,U]
    const float* b0     = (const float*)d_in[4];   // [U]
    const float* k1     = (const float*)d_in[5];   // [U,U]
    const float* rk1    = (const float*)d_in[6];   // [U,U]
    const float* b1     = (const float*)d_in[7];   // [U]
    const float* wd     = (const float*)d_in[8];   // [U,1]
    const float* bd     = (const float*)d_in[9];   // [1]
    float* out = (float*)d_out;                    // [B,1]

    // ---- workspace carve-up (all offsets multiple of 256B) ----
    char* ws = (char*)d_ws;
    size_t off = 0;
    __bf16* rk0t = (__bf16*)(ws + off); off += (size_t)U_ * U_ * 2;      // 512 KB
    __bf16* k1t  = (__bf16*)(ws + off); off += (size_t)U_ * U_ * 2;
    __bf16* rk1t = (__bf16*)(ws + off); off += (size_t)U_ * U_ * 2;
    float*  k0t  = (float*) (ws + off); off += (size_t)U_ * E_ * 4;      // 200 KB
    __bf16* hb0  = (__bf16*)(ws + off); off += (size_t)B_ * U_ * 2;      // 128 KB
    __bf16* hb1  = (__bf16*)(ws + off); off += (size_t)B_ * U_ * 2;
    __bf16* h0b  = (__bf16*)(ws + off); off += (size_t)B_ * U_ * 2;
    float*  xproj = (float*)(ws + off); off += (size_t)T_ * B_ * U_ * 4; // 20 MB

    // ---- one-time prep ----
    transpose_w_bf16<<<(U_ * U_) / 256, 256, 0, stream>>>(rk0, rk0t);
    transpose_w_bf16<<<(U_ * U_) / 256, 256, 0, stream>>>(k1,  k1t);
    transpose_w_bf16<<<(U_ * U_) / 256, 256, 0, stream>>>(rk1, rk1t);
    transpose_k0<<<(U_ * E_ + 255) / 256, 256, 0, stream>>>(k0, k0t);
    zero_bf16<<<(B_ * U_) / 256, 256, 0, stream>>>(hb0, B_ * U_);
    xproj_kernel<<<(T_ * B_ * U_) / 256, 256, 0, stream>>>(tokens, emb, k0t, b0, xproj);

    // ---- sequential recurrence: 2 WMMA GEMM kernels per step ----
    __bf16* hbuf[2] = { hb0, hb1 };
    int cur = 0;
    for (int t = 0; t < T_; ++t) {
        const float* xp_t = xproj + (size_t)t * B_ * U_;
        rnn_cell0<<<dim3(CELL_GRID), CELL_BLOCK, 0, stream>>>(hbuf[cur], rk0t, xp_t, h0b);
        rnn_cell1<<<dim3(CELL_GRID), CELL_BLOCK, 0, stream>>>(h0b, hbuf[cur], k1t, rk1t, b1,
                                                              hbuf[1 - cur]);
        cur = 1 - cur;
    }

    // ---- output head ----
    logits_kernel<<<1, B_, 0, stream>>>(hbuf[cur], wd, bd, out);
}
